// HoGT_22522808500913
// MI455X (gfx1250) — compile-verified, matrix operand order checked by reading, not run on hardware
//
#include <hip/hip_runtime.h>
#include <hip/hip_bf16.h>
#include <math.h>
#include <stdint.h>

typedef __attribute__((ext_vector_type(16))) __bf16 v16bf;
typedef __attribute__((ext_vector_type(8)))  __bf16 v8bf;
typedef __attribute__((ext_vector_type(2)))  __bf16 v2bf;
typedef __attribute__((ext_vector_type(8)))  float  v8f;

#define TILE_M 128
#define TILE_N 128
#define TILE_K 32
#define LDSS   48   // bf16 elements per LDS row (96B, 16B-aligned)
#define DH     512
#define NEG_SLOPE 0.2f

// ---------------------------------------------------------------------------
// f32 -> bf16 conversion (packed pairs)
// ---------------------------------------------------------------------------
__global__ void hogt_f32_to_bf16(const float* __restrict__ s,
                                 __bf16* __restrict__ d, long n2) {
    long i = (long)blockIdx.x * blockDim.x + threadIdx.x;
    if (i < n2) {
        float2 v = ((const float2*)s)[i];
        v2bf p;
        p[0] = (__bf16)v.x;
        p[1] = (__bf16)v.y;
        ((v2bf*)d)[i] = p;
    }
}

// W[K][N] f32  ->  Bt[N][K] bf16 (transposed, packed pairs along K)
__global__ void hogt_w_transpose_bf16(const float* __restrict__ W,
                                      __bf16* __restrict__ Bt, int K, int N) {
    long i = (long)blockIdx.x * blockDim.x + threadIdx.x;
    long total = (long)N * (K / 2);
    if (i >= total) return;
    int n = (int)(i / (K / 2));
    int k = (int)(i % (K / 2)) * 2;
    v2bf p;
    p[0] = (__bf16)W[(size_t)k * N + n];
    p[1] = (__bf16)W[(size_t)(k + 1) * N + n];
    *(v2bf*)&Bt[(size_t)n * K + k] = p;
}

// ---------------------------------------------------------------------------
// bf16 WMMA GEMM with async-to-LDS double-buffered staging.
//   A  [M][K] bf16 row-major, Bt [N][K] bf16 (B pre-transposed)
//   C  [M][N] f32 (+ bias[N])
// ---------------------------------------------------------------------------
__global__ __launch_bounds__(256)
void hogt_wmma_gemm_async(const __bf16* __restrict__ A,
                          const __bf16* __restrict__ Bt,
                          const float* __restrict__ bias,
                          float* __restrict__ C, int M, int N, int K) {
    __shared__ __bf16 As[2][TILE_M * LDSS];
    __shared__ __bf16 Bs[2][TILE_N * LDSS];

    const int tid    = threadIdx.x;
    const int lane   = tid & 31;
    const int laneLo = lane & 15;
    const int laneHi = lane >> 4;
    const int wave   = tid >> 5;
    const int wm     = wave & 3;           // 4 waves x 32 rows
    const int wn     = wave >> 2;          // 2 waves x 64 cols

    const int blockRow = blockIdx.y * TILE_M;
    const int blockCol = blockIdx.x * TILE_N;

    // async staging geometry: 4 lanes per 32-elem (64B) row, 64 rows/pass, 2 passes
    const int sr = tid >> 2;               // 0..63
    const int sc = (tid & 3) * 8;          // bf16 element offset: 0,8,16,24

    auto issue_tile = [&](int buf, int k0) {
#pragma unroll
        for (int p = 0; p < 2; ++p) {
            int rr = sr + p * 64;
            // ---- A tile row (guard M on the DMA itself; stale rows are discarded)
            {
                int grow = blockRow + rr;
                uint32_t la = (uint32_t)(uintptr_t)(&As[buf][rr * LDSS + sc]);
                const __bf16* gp = A + (size_t)grow * K + k0 + sc;
                if (grow < M)
                    asm volatile("global_load_async_to_lds_b128 %0, %1, off"
                                 :: "v"(la), "v"((uint64_t)(uintptr_t)gp)
                                 : "memory");
            }
            // ---- B tile row (Bt is [N][K], contiguous along K)
            {
                uint32_t lb = (uint32_t)(uintptr_t)(&Bs[buf][rr * LDSS + sc]);
                const __bf16* gq = Bt + (size_t)(blockCol + rr) * K + k0 + sc;
                asm volatile("global_load_async_to_lds_b128 %0, %1, off"
                             :: "v"(lb), "v"((uint64_t)(uintptr_t)gq)
                             : "memory");
            }
        }
    };

    v8f acc[2][4];
#pragma unroll
    for (int mi = 0; mi < 2; ++mi)
#pragma unroll
        for (int ni = 0; ni < 4; ++ni)
#pragma unroll
            for (int r = 0; r < 8; ++r) acc[mi][ni][r] = 0.0f;

    const int nT = K / TILE_K;
    issue_tile(0, 0);

    for (int t = 0; t < nT; ++t) {
        const int buf = t & 1;
        asm volatile("s_wait_asynccnt 0x0" ::: "memory");
        __syncthreads();                        // tile[buf] fully in LDS
        if (t + 1 < nT) issue_tile(buf ^ 1, (t + 1) * TILE_K);

        const int kbase = laneHi * 8;
        union frag_u { v16bf v; v8bf h[2]; };
        v16bf afrag[2], bfrag[4];
#pragma unroll
        for (int mi = 0; mi < 2; ++mi) {
            int row = wm * 32 + mi * 16 + laneLo;
            frag_u u;
            u.h[0] = *(const v8bf*)&As[buf][row * LDSS + kbase];
            u.h[1] = *(const v8bf*)&As[buf][row * LDSS + 16 + kbase];
            afrag[mi] = u.v;
        }
#pragma unroll
        for (int ni = 0; ni < 4; ++ni) {
            int col = wn * 64 + ni * 16 + laneLo;
            frag_u u;
            u.h[0] = *(const v8bf*)&Bs[buf][col * LDSS + kbase];
            u.h[1] = *(const v8bf*)&Bs[buf][col * LDSS + 16 + kbase];
            bfrag[ni] = u.v;
        }
#pragma unroll
        for (int mi = 0; mi < 2; ++mi)
#pragma unroll
            for (int ni = 0; ni < 4; ++ni)
                acc[mi][ni] = __builtin_amdgcn_wmma_f32_16x16x32_bf16(
                    false, afrag[mi], false, bfrag[ni],
                    (short)0, acc[mi][ni], false, false);
        // single barrier per iteration: frag reads retire (dscnt waits) before
        // each wave reaches the next iteration's barrier, so the async refill
        // of the other buffer cannot race them.
    }

    // epilogue: C/D layout VGPR r -> M = r + 8*laneHi, N = laneLo
#pragma unroll
    for (int mi = 0; mi < 2; ++mi)
#pragma unroll
        for (int ni = 0; ni < 4; ++ni) {
            int colBase = blockCol + wn * 64 + ni * 16 + laneLo;
            float bv = bias ? bias[colBase] : 0.0f;
#pragma unroll
            for (int r = 0; r < 8; ++r) {
                int row = blockRow + wm * 32 + mi * 16 + laneHi * 8 + r;
                if (row < M)
                    C[(size_t)row * N + colBase] = acc[mi][ni][r] + bv;
            }
        }
}

// ---------------------------------------------------------------------------
// el/er mat-vecs: one wave per node row
// ---------------------------------------------------------------------------
__global__ __launch_bounds__(256)
void hogt_rowdot2(const float* __restrict__ z, const float* __restrict__ al,
                  const float* __restrict__ ar, float* __restrict__ el,
                  float* __restrict__ er, int n) {
    int row  = blockIdx.x * 8 + (threadIdx.x >> 5);
    int lane = threadIdx.x & 31;
    if (row >= n) return;
    const float* zp = z + (size_t)row * DH;
    float sl = 0.f, sr = 0.f;
    for (int k = lane; k < DH; k += 32) {
        float v = zp[k];
        sl += v * al[k];
        sr += v * ar[k];
    }
#pragma unroll
    for (int off = 16; off > 0; off >>= 1) {
        sl += __shfl_down(sl, off, 32);
        sr += __shfl_down(sr, off, 32);
    }
    if (lane == 0) { el[row] = sl; er[row] = sr; }
}

__global__ void hogt_fill(float* __restrict__ p, float v, long n) {
    long i = (long)blockIdx.x * blockDim.x + threadIdx.x;
    if (i < n) p[i] = v;
}

__global__ void hogt_init_bias_rows(float* __restrict__ out,
                                    const float* __restrict__ b, long total) {
    long i = (long)blockIdx.x * blockDim.x + threadIdx.x;
    if (i < total) out[i] = b[i & (DH - 1)];
}

__global__ void hogt_copy(float* __restrict__ d, const float* __restrict__ s, long n) {
    long i = (long)blockIdx.x * blockDim.x + threadIdx.x;
    if (i < n) d[i] = s[i];
}

// e = leaky_relu(el[src]+er[dst]); segment max into m[dst] (ordered-int trick)
__global__ void hogt_edge_leaky_max(const float* __restrict__ el,
                                    const float* __restrict__ er,
                                    const int* __restrict__ src,
                                    const int* __restrict__ dst,
                                    float* __restrict__ eval,
                                    float* __restrict__ m, int E) {
    int i = blockIdx.x * blockDim.x + threadIdx.x;
    if (i >= E) return;
    float v = el[src[i]] + er[dst[i]];
    v = v > 0.f ? v : NEG_SLOPE * v;
    eval[i] = v;
    float* addr = &m[dst[i]];
    if (v >= 0.f) atomicMax((int*)addr, __float_as_int(v));
    else          atomicMin((unsigned int*)addr, __float_as_uint(v));
}

// ex = exp(e - m[dst]); denom[dst] += ex
__global__ void hogt_edge_exp_sum(float* __restrict__ eval,
                                  const float* __restrict__ m,
                                  const int* __restrict__ dst,
                                  float* __restrict__ denom, int E) {
    int i = blockIdx.x * blockDim.x + threadIdx.x;
    if (i >= E) return;
    float ex = __expf(eval[i] - m[dst[i]]);
    eval[i] = ex;
    atomicAdd(&denom[dst[i]], ex);
}

// out[dst] += (ex/denom[dst]) * z[src]   (one block per edge)
__global__ __launch_bounds__(256)
void hogt_gat_aggregate(const float* __restrict__ z, const float* __restrict__ eval,
                        const float* __restrict__ denom,
                        const int* __restrict__ src, const int* __restrict__ dst,
                        float* __restrict__ out) {
    int e = blockIdx.x;
    int s = src[e], d = dst[e];
    float coef = eval[e] / denom[d];
    const float* zs = z + (size_t)s * DH;
    float* od = out + (size_t)d * DH;
    for (int f = threadIdx.x; f < DH; f += blockDim.x)
        atomicAdd(&od[f], coef * zs[f]);
}

// znew[dst] += zprev[src]  (one block per edge)
__global__ __launch_bounds__(256)
void hogt_scatter_add(const float* __restrict__ zprev,
                      const int* __restrict__ src, const int* __restrict__ dst,
                      float* __restrict__ znew) {
    int e = blockIdx.x;
    int s = src[e], d = dst[e];
    const float* zs = zprev + (size_t)s * DH;
    float* zd = znew + (size_t)d * DH;
    for (int f = threadIdx.x; f < DH; f += blockDim.x)
        atomicAdd(&zd[f], zs[f]);
}

// ---------------------------------------------------------------------------
extern "C" void kernel_launch(void* const* d_in, const int* in_sizes, int n_in,
                              void* d_out, int out_size, void* d_ws, size_t ws_size,
                              hipStream_t stream) {
    const float* feat   = (const float*)d_in[0];
    const float* fc_W   = (const float*)d_in[1];
    const float* fc_b   = (const float*)d_in[2];
    const float* gat_W  = (const float*)d_in[3];
    const float* gat_al = (const float*)d_in[4];
    const float* gat_ar = (const float*)d_in[5];
    const float* gat_b  = (const float*)d_in[6];
    /* d_in[7] = beta: dead in the reference (it returns Z_prev) */
    const int*   src    = (const int*)d_in[8];
    const int*   dst    = (const int*)d_in[9];
    float*       out    = (float*)d_out;

    const int K0 = 768, D = DH;
    const int N  = in_sizes[0] / K0;    // 20000
    const int E  = in_sizes[8];         // 320000

    char* ws = (char*)d_ws;
    size_t matB = (size_t)N * D * sizeof(float);
    float*  bufH  = (float*)(ws);
    float*  bufZ  = (float*)(ws + matB);
    float*  bufO  = (float*)(ws + 2 * matB);
    float*  el    = (float*)(ws + 3 * matB);
    float*  er    = el + N;
    float*  mmax  = er + N;
    float*  denom = mmax + N;
    float*  eval  = denom + N;                       // E floats
    __bf16* Abf   = (__bf16*)(eval + E);             // N*K0 bf16 (covers N*D too)
    __bf16* Wt    = Abf + (size_t)N * K0;            // K0*D bf16 (pre-transposed W)

    long tot = (long)N * D;
    dim3 gemmGrid(D / TILE_N, (N + TILE_M - 1) / TILE_M);
    int  flatBlocks = (int)((tot + 255) / 256);
    int  edgeBlocks = (E + 255) / 256;
    int  nodeBlocks = (N + 255) / 256;

    // ---- h = feat @ fc_W + fc_b (bf16 WMMA, async staging)
    {
        long n2 = (long)N * K0 / 2;
        hogt_f32_to_bf16<<<(int)((n2 + 255) / 256), 256, 0, stream>>>(feat, Abf, n2);
        long wtot = (long)D * (K0 / 2);
        hogt_w_transpose_bf16<<<(int)((wtot + 255) / 256), 256, 0, stream>>>(fc_W, Wt, K0, D);
        hogt_wmma_gemm_async<<<gemmGrid, 256, 0, stream>>>(Abf, Wt, fc_b, bufH, N, D, K0);
    }

    float* h = bufH;
    float* o = bufO;
    for (int l = 0; l < 4; ++l) {
        const float* Wl  = gat_W + (size_t)l * D * D;
        const float* all = gat_al + (size_t)l * D;
        const float* arl = gat_ar + (size_t)l * D;
        const float* bl  = gat_b + (size_t)l * D;

        long n2 = tot / 2;
        hogt_f32_to_bf16<<<(int)((n2 + 255) / 256), 256, 0, stream>>>(h, Abf, n2);
        long wtot = (long)D * (D / 2);
        hogt_w_transpose_bf16<<<(int)((wtot + 255) / 256), 256, 0, stream>>>(Wl, Wt, D, D);
        hogt_wmma_gemm_async<<<gemmGrid, 256, 0, stream>>>(Abf, Wt, nullptr, bufZ, N, D, D);

        hogt_rowdot2<<<(N + 7) / 8, 256, 0, stream>>>(bufZ, all, arl, el, er, N);
        hogt_fill<<<nodeBlocks, 256, 0, stream>>>(mmax, -INFINITY, N);
        hogt_fill<<<nodeBlocks, 256, 0, stream>>>(denom, 0.0f, N);
        hogt_edge_leaky_max<<<edgeBlocks, 256, 0, stream>>>(el, er, src, dst, eval, mmax, E);
        hogt_edge_exp_sum<<<edgeBlocks, 256, 0, stream>>>(eval, mmax, dst, denom, E);
        hogt_init_bias_rows<<<flatBlocks, 256, 0, stream>>>(o, bl, tot);
        hogt_gat_aggregate<<<E, 256, 0, stream>>>(bufZ, eval, denom, src, dst, o);

        float* t = h; h = o; o = t;     // h -> this layer's output
    }
    // after 4 swaps: h == bufH holds GAT output; bufZ, bufO free

    // 3 propagation hops of A_hat = A + I (beta is dead; output = A_hat^3 @ h)
    hogt_copy<<<flatBlocks, 256, 0, stream>>>(bufZ, h, tot);
    hogt_scatter_add<<<E, 256, 0, stream>>>(h, src, dst, bufZ);
    hogt_copy<<<flatBlocks, 256, 0, stream>>>(bufO, bufZ, tot);
    hogt_scatter_add<<<E, 256, 0, stream>>>(bufZ, src, dst, bufO);
    hogt_copy<<<flatBlocks, 256, 0, stream>>>(out, bufO, tot);
    hogt_scatter_add<<<E, 256, 0, stream>>>(bufO, src, dst, out);
}
